// PAM_Module_85615878078539
// MI455X (gfx1250) — matrix operand order repaired and checked by reading, not compiled, not run on hardware
//
#include <hip/hip_runtime.h>
#include <hip/hip_bf16.h>

typedef __attribute__((ext_vector_type(16))) __bf16  v16bf;
typedef __attribute__((ext_vector_type(8)))  __bf16  v8bf;
typedef __attribute__((ext_vector_type(8)))  float   v8f;

union BF16x16 { v16bf v; v8bf h[2]; unsigned short s[16]; };
union BF16x8  { v8bf  v; unsigned short s[8]; };

__device__ __forceinline__ unsigned short f2bf(float f) {
  __bf16 h = (__bf16)f;                       // native v_cvt, RNE
  return __builtin_bit_cast(unsigned short, h);
}

__device__ __forceinline__ v8f wmma_bf16(v16bf a, v16bf b, v8f c) {
  return __builtin_amdgcn_wmma_f32_16x16x32_bf16(
      /*neg_a=*/false, a, /*neg_b=*/false, b,
      /*c_mod=*/(short)0, c, /*reuse_a=*/false, /*reuse_b=*/false);
}

// generic (flat) shared-memory pointer -> raw LDS byte offset (addr[31:0])
__device__ __forceinline__ unsigned lds_off_of(const void* p) {
  return (unsigned)(uintptr_t)p;
}

// gfx1250 async global->LDS DMA (ASYNCcnt-tracked), 16 bytes per lane
__device__ __forceinline__ void async_g2l_b128(unsigned lds_off, const void* g) {
  asm volatile("global_load_async_to_lds_b128 %0, %1, off"
               :: "v"(lds_off), "v"(g) : "memory");
}
__device__ __forceinline__ void wait_async0() {
  asm volatile("s_wait_asynccnt 0" ::: "memory");
}

#define CDIM 512
#define NDIM 4096
#define BDIM 8
#define HKDIM 64    // head dim for Q/K
#define VDDIM 512   // value dim
#define NQ 32       // query rows per attention block
#define NKV 128     // key block (2x deeper WMMA clauses, half the softmax/barrier overhead)

// -------- Projection GEMM: out_bf16(b, M, N) = epi( W(MxC) @ x(b, C, N) ) --------
// MODE 0: + bias(p0)   (Q/K).  MODE 1: relu(BN(p0..p3))  (V).
// 128 threads = 4 waves; tile 16(M) x 64(N); double-buffered async staging of x/W.
template <int MODE>
__global__ __launch_bounds__(128) void proj_kernel(
    const float* __restrict__ Wm,
    const float* __restrict__ p0, const float* __restrict__ p1,
    const float* __restrict__ p2, const float* __restrict__ p3,
    const float* __restrict__ x,
    unsigned short* __restrict__ out, int M) {
  const int tid  = threadIdx.x;
  const int lane = tid & 31;
  const int wave = tid >> 5;             // 0..3 -> N sub-tile
  const int tn0  = blockIdx.x * 64;
  const int tm   = blockIdx.y * 16;
  const int b    = blockIdx.z;
  const float* xb = x + (size_t)b * CDIM * NDIM;

  __shared__ __attribute__((aligned(16))) float Xsf[2][32 * 64];  // [c][n]  16 KB
  __shared__ __attribute__((aligned(16))) float Wsf[2][16 * 32];  // [m][k]   4 KB
  const unsigned xs0 = lds_off_of(Xsf);
  const unsigned ws0 = lds_off_of(Wsf);

  auto issue_stage = [&](int k0, int buf) {
    unsigned xsb = xs0 + (unsigned)buf * (32 * 64 * 4);
    unsigned wsb = ws0 + (unsigned)buf * (16 * 32 * 4);
#pragma unroll
    for (int kk = 0; kk < 4; ++kk) {               // x tile: 512 x 16B chunks
      int c = tid + kk * 128;
      int r = c >> 4, cc = (c & 15) * 4;
      async_g2l_b128(xsb + (unsigned)(r * 64 + cc) * 4u,
                     xb + (size_t)(k0 + r) * NDIM + tn0 + cc);
    }
    int r = tid >> 3, cc = (tid & 7) * 4;          // W tile: 128 x 16B chunks
    async_g2l_b128(wsb + (unsigned)(r * 32 + cc) * 4u,
                   Wm + (size_t)(tm + r) * CDIM + k0 + cc);
  };

  v8f acc = {};
  const int mA  = lane & 15, kbA = (lane < 16) ? 0 : 16;
  const int nB  = lane & 15, kbB = (lane < 16) ? 0 : 16;

  issue_stage(0, 0);
  for (int it = 0; it < CDIM / 32; ++it) {
    wait_async0();
    __syncthreads();                               // stage(it) visible to all waves
    if (it + 1 < CDIM / 32) issue_stage((it + 1) * 32, (it + 1) & 1);

    const float* Wt = Wsf[it & 1];
    const float* Xt = Xsf[it & 1];
    BF16x16 a, bf;
#pragma unroll
    for (int j = 0; j < 8; ++j) {
      a.s[j]     = f2bf(Wt[mA * 32 + kbA + j]);
      a.s[8 + j] = f2bf(Wt[mA * 32 + kbA + 8 + j]);
    }
#pragma unroll
    for (int j = 0; j < 16; ++j)
      bf.s[j] = f2bf(Xt[(kbB + j) * 64 + wave * 16 + nB]);
    acc = wmma_bf16(a.v, bf.v, acc);
  }

  const int mb = (lane < 16) ? 0 : 8;
#pragma unroll
  for (int r = 0; r < 8; ++r) {
    int m = tm + mb + r;
    float sc, sh;
    if (MODE == 0) { sc = 1.f; sh = p0[m]; }
    else {
      sc = p0[m] * rsqrtf(p3[m] + 1e-5f);
      sh = p1[m] - p2[m] * sc;
    }
    float vv = acc[r] * sc + sh;
    if (MODE == 1) vv = fmaxf(vv, 0.f);
    out[((size_t)b * M + m) * NDIM + tn0 + wave * 16 + nB] = f2bf(vv);
  }
}

// -------- Fused flash attention: out(b, V, n) = V_relu @ softmax(Q K)^T --------
__global__ __launch_bounds__(256)
void pam_attn_kernel(const unsigned short* __restrict__ qbuf,   // (B, 64, N) bf16
                     const unsigned short* __restrict__ kbuf,   // (B, 64, N) bf16
                     const unsigned short* __restrict__ vbuf,   // (B, 512, N) bf16
                     float* __restrict__ out) {                 // (B, 512, N) f32
  const int b    = blockIdx.y;
  const int q0   = blockIdx.x * NQ;
  const int tid  = threadIdx.x;
  const int lane = tid & 31;
  const int wave = tid >> 5;              // 0..7

  __shared__ __attribute__((aligned(16))) unsigned short Qt[NQ * HKDIM];       //  4 KB
  __shared__ __attribute__((aligned(16))) unsigned short Ksb[2][HKDIM * NKV];  // 32 KB
  __shared__ __attribute__((aligned(16))) float          Ss[NQ * NKV];         // 16 KB
  __shared__ __attribute__((aligned(16))) unsigned short Ps[NQ * NKV];         //  8 KB
  __shared__ float rowmax[NQ], rowsum[NQ], alphas[NQ];
  const unsigned ks0 = lds_off_of(Ksb);

  const unsigned short* qb = qbuf + (size_t)b * HKDIM * NDIM;
  const unsigned short* kb = kbuf + (size_t)b * HKDIM * NDIM;
  const unsigned short* vb = vbuf + (size_t)b * VDDIM * NDIM;

  auto issueK = [&](int kblk, int buf) {          // 64x128 bf16 tile: 1024 x 16B chunks
    unsigned base = ks0 + (unsigned)buf * (HKDIM * NKV * 2);
#pragma unroll
    for (int kk = 0; kk < 4; ++kk) {
      int c = tid + kk * 256;
      int r = c >> 4, cc = (c & 15) * 8;
      async_g2l_b128(base + (unsigned)(r * NKV + cc) * 2u,
                     kb + (size_t)r * NDIM + kblk + cc);
    }
  };

  { // stage Q transposed: Qt[qrow][head]  (strided global reads, once per block)
    int m = tid >> 3, h0 = (tid & 7) * 8;
#pragma unroll
    for (int j = 0; j < 8; ++j)
      Qt[m * HKDIM + h0 + j] = qb[(size_t)(h0 + j) * NDIM + q0 + m];
  }
  if (tid < NQ) { rowmax[tid] = -1e30f; rowsum[tid] = 0.f; }
  issueK(0, 0);
  __syncthreads();

  // Q A-fragments (held in registers for all key blocks) — contiguous vector loads
  BF16x16 qfrag[2][2];
  {
    int m = lane & 15, kbA = (lane < 16) ? 0 : 16;
#pragma unroll
    for (int mt = 0; mt < 2; ++mt)
#pragma unroll
      for (int ks = 0; ks < 2; ++ks) {
        const unsigned short* qp = &Qt[(mt * 16 + m) * HKDIM + ks * 32];
        qfrag[mt][ks].h[0] = *(const v8bf*)(qp + kbA);
        qfrag[mt][ks].h[1] = *(const v8bf*)(qp + 8 + kbA);
      }
  }

  v8f acc[2][4];   // [qrow tile][vcol subtile]; this wave owns vcols [wave*64, wave*64+64)
  { v8f z = {}; for (int mt = 0; mt < 2; ++mt) for (int vt = 0; vt < 4; ++vt) acc[mt][vt] = z; }
  const int vcol0 = wave * 64;

  const int NITER = NDIM / NKV;   // 32
  for (int it = 0; it < NITER; ++it) {
    const int kblk = it * NKV;
    wait_async0();
    __syncthreads();   // K(it) landed everywhere; previous iteration fully consumed

    if (it + 1 < NITER) issueK(kblk + NKV, (it + 1) & 1);   // overlap DMA with compute
    if (kblk + 2 * NKV < NDIM) {                            // prefetch K two blocks out
      int r = tid >> 3, cc = (tid & 7) * 16;
      __builtin_prefetch(kb + (size_t)r * NDIM + kblk + 2 * NKV + cc, 0, 1);
      __builtin_prefetch(kb + (size_t)(r + 32) * NDIM + kblk + 2 * NKV + cc, 0, 1);
    }
    if (kblk + NKV < NDIM) {                                // prefetch this wave's next V rows
      const unsigned short* vpn = vb + (size_t)(vcol0 + lane) * NDIM + kblk + NKV;
      __builtin_prefetch(vpn, 0, 1);
      __builtin_prefetch(vpn + 64, 0, 1);
      __builtin_prefetch(vpn + (size_t)32 * NDIM, 0, 1);
      __builtin_prefetch(vpn + (size_t)32 * NDIM + 64, 0, 1);
    }

    { // S = Q @ K : two 16x16 tiles per wave (2 WMMA k-steps each over head=64)
      const unsigned short* Ks = Ksb[it & 1];
      int n = lane & 15, kbB = (lane < 16) ? 0 : 16;
      int mb = (lane < 16) ? 0 : 8;
#pragma unroll
      for (int tt = 0; tt < 2; ++tt) {
        int tile = wave * 2 + tt;            // 16 tiles: 2 (qrow) x 8 (key)
        int smt = tile >> 3, snt = tile & 7;
        v8f s = {};
#pragma unroll
        for (int ks = 0; ks < 2; ++ks) {
          BF16x16 bf;
#pragma unroll
          for (int j = 0; j < 16; ++j)
            bf.s[j] = Ks[(ks * 32 + kbB + j) * NKV + snt * 16 + n];
          s = wmma_bf16(qfrag[smt][ks].v, bf.v, s);
        }
#pragma unroll
        for (int r = 0; r < 8; ++r) Ss[(smt * 16 + mb + r) * NKV + snt * 16 + n] = s[r];
      }
    }
    __syncthreads();

    { // parallel online softmax: 8 threads per row x 16 cols, shfl-xor reductions
      int r  = tid >> 3;            // 0..31
      int c0 = (tid & 7) * 16;
      float sv[16];
      float mloc = -1e30f;
#pragma unroll
      for (int j = 0; j < 16; ++j) {
        sv[j] = Ss[r * NKV + c0 + j];
        mloc = fmaxf(mloc, sv[j]);
      }
#pragma unroll
      for (int m = 1; m <= 4; m <<= 1) mloc = fmaxf(mloc, __shfl_xor(mloc, m));
      float mold = rowmax[r];
      float mnew = fmaxf(mold, mloc);
      float al   = __expf(mold - mnew);
      BF16x8 pk0, pk1;
      float ps = 0.f;
#pragma unroll
      for (int j = 0; j < 8; ++j) {
        float p = __expf(sv[j] - mnew);
        pk0.s[j] = f2bf(p);
        ps += p;
      }
#pragma unroll
      for (int j = 0; j < 8; ++j) {
        float p = __expf(sv[8 + j] - mnew);
        pk1.s[j] = f2bf(p);
        ps += p;
      }
      *(v8bf*)&Ps[r * NKV + c0]     = pk0.v;   // 16B vector stores
      *(v8bf*)&Ps[r * NKV + c0 + 8] = pk1.v;
#pragma unroll
      for (int m = 1; m <= 4; m <<= 1) ps += __shfl_xor(ps, m);
      if ((tid & 7) == 0) {
        rowmax[r] = mnew;
        rowsum[r] = rowsum[r] * al + ps;
        alphas[r] = al;
      }
    }
    __syncthreads();

    { // rescale accumulators, then O += P @ V^T  (32 WMMAs per wave per block)
      int mb = (lane < 16) ? 0 : 8;
#pragma unroll
      for (int mt = 0; mt < 2; ++mt)
#pragma unroll
        for (int r = 0; r < 8; ++r) {
          float al = alphas[mt * 16 + mb + r];
#pragma unroll
          for (int vt = 0; vt < 4; ++vt) acc[mt][vt][r] *= al;
        }

      int m = lane & 15, kbA = (lane < 16) ? 0 : 16;
      int n = lane & 15, kbB = (lane < 16) ? 0 : 16;
#pragma unroll
      for (int ks = 0; ks < 4; ++ks) {      // 4 x 32-key steps
        BF16x16 pf[2];
#pragma unroll
        for (int mt = 0; mt < 2; ++mt) {
          const unsigned short* pp = &Ps[(mt * 16 + m) * NKV + ks * 32];
          pf[mt].h[0] = *(const v8bf*)(pp + kbA);
          pf[mt].h[1] = *(const v8bf*)(pp + 8 + kbA);
        }
#pragma unroll
        for (int vt = 0; vt < 4; ++vt) {
          // V B-fragment: 16 contiguous bf16 per lane, 32B aligned -> vector load
          const unsigned short* vp =
              vb + (size_t)(vcol0 + vt * 16 + n) * NDIM + kblk + ks * 32 + kbB;
          BF16x16 vf;
          vf.v = *(const v16bf*)vp;
#pragma unroll
          for (int mt = 0; mt < 2; ++mt)
            acc[mt][vt] = wmma_bf16(pf[mt].v, vf.v, acc[mt][vt]);
        }
      }
    }
    // next iteration's top barrier orders Ps/Ks reuse
  }

  { // normalize by rowsum and store f32 output (B, 512, N)
    int n = lane & 15, mb = (lane < 16) ? 0 : 8;
#pragma unroll
    for (int mt = 0; mt < 2; ++mt) {
#pragma unroll
      for (int r = 0; r < 8; ++r) {
        float inv = 1.f / rowsum[mt * 16 + mb + r];
        int qq = q0 + mt * 16 + mb + r;
#pragma unroll
        for (int vt = 0; vt < 4; ++vt) {
          int vcol = vcol0 + vt * 16 + n;
          out[((size_t)b * VDDIM + vcol) * NDIM + qq] = acc[mt][vt][r] * inv;
        }
      }
    }
  }
}

extern "C" void kernel_launch(void* const* d_in, const int* in_sizes, int n_in,
                              void* d_out, int out_size, void* d_ws, size_t ws_size,
                              hipStream_t stream) {
  const float* x     = (const float*)d_in[0];
  const float* Wq    = (const float*)d_in[1];
  const float* bq    = (const float*)d_in[2];
  const float* Wk    = (const float*)d_in[3];
  const float* bk    = (const float*)d_in[4];
  const float* Wv    = (const float*)d_in[5];
  const float* gamma = (const float*)d_in[6];
  const float* beta  = (const float*)d_in[7];
  const float* mean  = (const float*)d_in[8];
  const float* var   = (const float*)d_in[9];
  float* out = (float*)d_out;

  // bf16 workspace: Q(8,64,4096) K(8,64,4096) V(8,512,4096) = 40 MB
  unsigned short* qbuf = (unsigned short*)d_ws;
  unsigned short* kbuf = qbuf + (size_t)BDIM * HKDIM * NDIM;
  unsigned short* vbuf = kbuf + (size_t)BDIM * HKDIM * NDIM;

  dim3 blk(128);
  proj_kernel<0><<<dim3(NDIM / 64, HKDIM / 16, BDIM), blk, 0, stream>>>(
      Wq, bq, nullptr, nullptr, nullptr, x, qbuf, HKDIM);
  proj_kernel<0><<<dim3(NDIM / 64, HKDIM / 16, BDIM), blk, 0, stream>>>(
      Wk, bk, nullptr, nullptr, nullptr, x, kbuf, HKDIM);
  proj_kernel<1><<<dim3(NDIM / 64, VDDIM / 16, BDIM), blk, 0, stream>>>(
      Wv, gamma, beta, mean, var, x, vbuf, VDDIM);

  pam_attn_kernel<<<dim3(NDIM / NQ, BDIM), dim3(256), 0, stream>>>(qbuf, kbuf, vbuf, out);
}
